// VectorQuantizer_43817256353929
// MI455X (gfx1250) — matrix operand order repaired and checked by reading, not compiled, not run on hardware
//
#include <hip/hip_runtime.h>

typedef __attribute__((ext_vector_type(16))) __bf16 v16bf;
typedef __attribute__((ext_vector_type(2)))  __bf16 bf16x2v;
typedef __attribute__((ext_vector_type(8)))  float  v8f;

#define D_DIM 64
#define K_CODES 512
#define N_ROWS (64*64*64)      // 262144 flattened rows
#define ROWS_PER_WAVE 32       // two 16-row WMMA tiles per wave
#define ROWS_PER_BLOCK 256     // 8 waves x 32 rows
#define THREADS 256
#define LDS_EMB_BYTES (K_CODES * D_DIM * 2)            // 64 KB bf16 codebook
#define LDS_BYTES (LDS_EMB_BYTES + K_CODES * 4)        // + 2 KB negHalf table

__device__ __forceinline__ unsigned pack_bf16x2(float lo, float hi) {
    // native RNE f32->bf16 converts; backend can fuse into packed cvt
    bf16x2v p;
    p.x = (__bf16)lo;
    p.y = (__bf16)hi;
    return __builtin_bit_cast(unsigned, p);
}

// ---- prep: zero the loss accumulator ----
__global__ void vq_prep(float* __restrict__ ws) { ws[0] = 0.0f; }

// ---- main: WMMA distance matmul + argmin + gather + loss partials ----
__global__ __launch_bounds__(THREADS) void vq_main(const float* __restrict__ in,
                                                   const float* __restrict__ emb,
                                                   float* __restrict__ out,
                                                   float* __restrict__ ws) {
    extern __shared__ unsigned char smemRaw[];
    unsigned* embLds     = (unsigned*)smemRaw;                       // [512][32] bf16x2 row-major
    float*    negHalfLds = (float*)(smemRaw + LDS_EMB_BYTES);        // [512] = -0.5*||e_k||^2

    const int tid = threadIdx.x;

    // Stage codebook into LDS as bf16 and compute -0.5*||e||^2 in the same pass.
    for (int r = tid; r < K_CODES; r += THREADS) {
        const float4* src = (const float4*)(emb + r * D_DIM);
        uint4* dst = (uint4*)(embLds + r * 32);
        float s = 0.0f;
        #pragma unroll
        for (int j = 0; j < 8; ++j) {
            float4 v0 = src[2*j];
            float4 v1 = src[2*j + 1];
            s += v0.x*v0.x + v0.y*v0.y + v0.z*v0.z + v0.w*v0.w;
            s += v1.x*v1.x + v1.y*v1.y + v1.z*v1.z + v1.w*v1.w;
            uint4 p;
            p.x = pack_bf16x2(v0.x, v0.y);
            p.y = pack_bf16x2(v0.z, v0.w);
            p.z = pack_bf16x2(v1.x, v1.y);
            p.w = pack_bf16x2(v1.z, v1.w);
            dst[j] = p;
        }
        negHalfLds[r] = -0.5f * s;
    }
    __syncthreads();

    const int wave = tid >> 5;
    const int lane = tid & 31;
    const int m16  = lane & 15;   // A row within tile / B column within tile
    const int hi   = lane >> 4;   // lane-half selector per ISA VGPR layouts

    const int rowBase = blockIdx.x * ROWS_PER_BLOCK + wave * ROWS_PER_WAVE;

    // ---- Load two 16x64 A tiles -> four 16x32 bf16 operands, ISA A-layout:
    // lane m16 holds row m16; chunk c: vgpr0-3 = K c*32+hi*8+{0..7}, vgpr4-7 = K c*32+16+hi*8+{0..7}
    union AOp { v16bf v; uint4 q[2]; } aOp[2][2];   // [row-subtile][k-chunk]
    #pragma unroll
    for (int s = 0; s < 2; ++s) {
        const float4* f = (const float4*)(in + (rowBase + s*16 + m16) * D_DIM);
        #pragma unroll
        for (int c = 0; c < 2; ++c) {
            float4 g0a = f[c*8 + hi*2];
            float4 g0b = f[c*8 + hi*2 + 1];
            float4 g1a = f[c*8 + 4 + hi*2];
            float4 g1b = f[c*8 + 4 + hi*2 + 1];
            aOp[s][c].q[0] = make_uint4(pack_bf16x2(g0a.x, g0a.y), pack_bf16x2(g0a.z, g0a.w),
                                        pack_bf16x2(g0b.x, g0b.y), pack_bf16x2(g0b.z, g0b.w));
            aOp[s][c].q[1] = make_uint4(pack_bf16x2(g1a.x, g1a.y), pack_bf16x2(g1a.z, g1a.w),
                                        pack_bf16x2(g1b.x, g1b.y), pack_bf16x2(g1b.z, g1b.w));
        }
    }

    float bestVal[2][8];
    int   bestIdx[2][8];
    #pragma unroll
    for (int s = 0; s < 2; ++s)
        #pragma unroll
        for (int r = 0; r < 8; ++r) { bestVal[s][r] = -3.4e38f; bestIdx[s][r] = 0; }

    const v8f vzero = { 0.f, 0.f, 0.f, 0.f, 0.f, 0.f, 0.f, 0.f };

    // ---- Sweep 32 code tiles; score = x.e - 0.5||e||^2, argmax == argmin distance
    for (int t = 0; t < 32; ++t) {
        const int col = t * 16 + m16;            // this lane's code index within tile
        const float c0 = negHalfLds[col];        // ds load (same wait domain as B)

        // B-layout (32x16 bf16): lane holds column m16, per chunk K = hi*8+{0..7}, 16+hi*8+{0..7}
        const unsigned* brow = embLds + col * 32;
        union BOp { v16bf v; uint4 q[2]; } b0, b1;
        b0.q[0] = *(const uint4*)(brow + hi*4);
        b0.q[1] = *(const uint4*)(brow + 8 + hi*4);
        b1.q[0] = *(const uint4*)(brow + 16 + hi*4);
        b1.q[1] = *(const uint4*)(brow + 24 + hi*4);

        v8f acc0 = __builtin_amdgcn_wmma_f32_16x16x32_bf16(false, aOp[0][0].v, false, b0.v,
                                                           (short)0, vzero, false, false);
        v8f acc1 = __builtin_amdgcn_wmma_f32_16x16x32_bf16(false, aOp[1][0].v, false, b0.v,
                                                           (short)0, vzero, false, false);
        acc0 = __builtin_amdgcn_wmma_f32_16x16x32_bf16(false, aOp[0][1].v, false, b1.v,
                                                       (short)0, acc0, false, false);
        acc1 = __builtin_amdgcn_wmma_f32_16x16x32_bf16(false, aOp[1][1].v, false, b1.v,
                                                       (short)0, acc1, false, false);

        #pragma unroll
        for (int r = 0; r < 8; ++r) {
            float v0 = acc0[r] + c0;
            float v1 = acc1[r] + c0;
            if (v0 > bestVal[0][r]) { bestVal[0][r] = v0; bestIdx[0][r] = col; }
            if (v1 > bestVal[1][r]) { bestVal[1][r] = v1; bestIdx[1][r] = col; }
        }
    }

    // ---- argmax reduction across the 16 lanes sharing each C row (xor 1,2,4,8 keeps hi fixed)
    // Branchless tie-break (bitwise &,| -- no EXEC-mask short-circuit code).
    #pragma unroll
    for (int msk = 1; msk < 16; msk <<= 1) {
        #pragma unroll
        for (int s = 0; s < 2; ++s) {
            #pragma unroll
            for (int r = 0; r < 8; ++r) {
                float ov = __shfl_xor(bestVal[s][r], msk, 32);
                int   oi = __shfl_xor(bestIdx[s][r], msk, 32);
                bool take = (ov > bestVal[s][r]) |
                            ((ov == bestVal[s][r]) & (oi < bestIdx[s][r]));
                bestVal[s][r] = take ? ov : bestVal[s][r];
                bestIdx[s][r] = take ? oi : bestIdx[s][r];
            }
        }
    }

    // ---- gather fp32 embedding rows, write output, accumulate squared error
    float lossAcc = 0.0f;
    const float* inBase  = in  + rowBase * D_DIM;
    float*       outBase = out + rowBase * D_DIM;
    #pragma unroll
    for (int m = 0; m < ROWS_PER_WAVE; ++m) {
        const int s  = m >> 4;          // which 16-row subtile
        const int lm = m & 15;          // row within subtile
        // winning index is uniform across the owning 16-lane half: read into an SGPR
        int idx = __builtin_amdgcn_readlane(bestIdx[s][lm & 7], (lm >> 3) * 16);
        float2 q = *(const float2*)(emb    + idx * D_DIM + lane * 2);
        float2 x = *(const float2*)(inBase + m   * D_DIM + lane * 2);
        *(float2*)(outBase + m * D_DIM + lane * 2) = q;
        float dx = x.x - q.x, dy = x.y - q.y;
        lossAcc += dx * dx + dy * dy;
    }

    #pragma unroll
    for (int msk = 16; msk >= 1; msk >>= 1) lossAcc += __shfl_xor(lossAcc, msk, 32);
    if (lane == 0) atomicAdd(ws, lossAcc);
}

// ---- finalize the two scalar losses ----
__global__ void vq_final(const float* __restrict__ ws, float* __restrict__ out) {
    float mse = ws[0] / (float)(N_ROWS * D_DIM);
    out[(size_t)N_ROWS * D_DIM]     = mse;          // codebook_loss
    out[(size_t)N_ROWS * D_DIM + 1] = 0.25f * mse;  // commitment_loss
}

extern "C" void kernel_launch(void* const* d_in, const int* in_sizes, int n_in,
                              void* d_out, int out_size, void* d_ws, size_t ws_size,
                              hipStream_t stream) {
    const float* in  = (const float*)d_in[0];   // (64,64,64,64) f32
    const float* emb = (const float*)d_in[1];   // (512,64) f32
    float* out = (float*)d_out;
    float* ws  = (float*)d_ws;

    vq_prep<<<1, 1, 0, stream>>>(ws);
    vq_main<<<N_ROWS / ROWS_PER_BLOCK, THREADS, LDS_BYTES, stream>>>(in, emb, out, ws);
    vq_final<<<1, 1, 0, stream>>>(ws, out);
}
